// DBNN_59545426591797
// MI455X (gfx1250) — matrix operand order repaired and checked by reading, not compiled, not run on hardware
//
#include <hip/hip_runtime.h>

// MI455X / gfx1250, wave32. WMMA f16 path for the bilinear form,
// LDS-staged B-fragments, prefetch for the next stage.

typedef __attribute__((ext_vector_type(16))) _Float16 v16h;
typedef __attribute__((ext_vector_type(8)))  _Float16 v8h;
typedef __attribute__((ext_vector_type(8)))  float    v8f;

#define B_ 16
#define N_ 256
#define T_ 2048
#define THRES_ (-55.0f)
#define BIAS_ (-70.0f)
#define RESET_AMP_ 15.0f

// ---------------------------------------------------------------------------
// Kernel 1: depthwise causal conv as two IIR recurrences.
//   kern[t] = omega*(a1^t - a2^t),  a1=exp(-1/tau_d), a2=exp(-1/tau_r-1/tau_d)
//   E[t] = a*E[t-1] + x[t]  =>  y[t] = omega*(E1[t]-E2[t])
// One block per batch, one thread per channel n. Output written TRANSPOSED
// to (b, t, n) in f16 so lane=n stores are coalesced and WMMA-ready.
// ---------------------------------------------------------------------------
__global__ void dbnn_iir_conv(const float* __restrict__ x,
                              const float* __restrict__ tau_rise,
                              const float* __restrict__ tau_decay,
                              const float* __restrict__ omega,
                              _Float16* __restrict__ yp16) {
    const int b = blockIdx.x;
    const int n = threadIdx.x;
    const float invR = 1.0f / tau_rise[n];
    const float invD = 1.0f / tau_decay[n];
    const float a1 = __expf(-invD);
    const float a2 = __expf(-(invR + invD));
    const float om = omega[n];

    const float*  xr = x + ((size_t)b * N_ + n) * T_;
    _Float16*     yo = yp16 + (size_t)b * T_ * N_ + n;

    float E1 = 0.0f, E2 = 0.0f;
    for (int t = 0; t < T_; ++t) {
        const float xv = xr[t];
        E1 = fmaf(a1, E1, xv);
        E2 = fmaf(a2, E2, xv);
        yo[(size_t)t * N_] = (_Float16)(om * (E1 - E2));
    }
}

// ---------------------------------------------------------------------------
// Kernel 2: pack Wm = W[0] (diagonal zeroed) into WMMA B-fragment layout
// (16-bit B 32x16, contiguous k order, same as A): lane l holds column j0 + (l&15), K-rows
// 16*(l>>4) + e, 2 halves/VGPR. Tile index = jt*8 + kc8 (jt OUTERMOST so a
// jt-half [8 tiles of 16 cols] is one contiguous 64KB region for LDS staging).
// ---------------------------------------------------------------------------
__global__ void dbnn_pack_w(const float* __restrict__ W,
                            _Float16* __restrict__ wpack) {
    const int tile = blockIdx.x;          // 0..127
    const int jt   = tile >> 3;           // 0..15 (outermost)
    const int kc8  = tile & 7;            // 0..7  (K-chunk of 32)
    const int lane = threadIdx.x;         // 0..31
    const int h    = lane >> 4;
    const int j    = jt * 16 + (lane & 15);

    _Float16* dst = wpack + ((size_t)tile * 32 + lane) * 16;
#pragma unroll
    for (int e = 0; e < 16; ++e) {
        const int k = kc8 * 32 + 16 * h + e;
        float w = W[(size_t)k * N_ + j];
        if (k == j) w = 0.0f;             // Wm = W * (1 - eye)
        dst[e] = (_Float16)w;
    }
}

// ---------------------------------------------------------------------------
// Kernel 3: bilinear + linear term via WMMA, B-fragments staged in LDS.
//   op_raw[b,t] = sum_j (Z[t,j] + 1) * Yp[t,j] + BIAS,  Z = Yp @ Wm
// One wave per 16-row t-tile; block = 8 waves. Packed W (128KB) is consumed
// in two 64KB stages: cooperative global->LDS copy, barrier, then 8 column-
// tiles x 8 WMMAs each fed by ds_load_b128. Cuts L2 W-traffic 8x vs. per-
// wave global loads. All control flow is wave-uniform (EXEC all-1s at WMMA).
// ---------------------------------------------------------------------------
__global__ __launch_bounds__(256) void
dbnn_bilinear_wmma(const _Float16* __restrict__ yp16,
                   const _Float16* __restrict__ wpack,
                   float* __restrict__ opraw) {
    __shared__ __align__(16) _Float16 wlds[32768];     // 64 KB: one jt-half

    const int b      = blockIdx.x;
    const int wave   = threadIdx.x >> 5;
    const int lane   = threadIdx.x & 31;
    const int lane16 = lane & 15;
    const int h      = lane >> 4;
    const int t0     = (blockIdx.y * 8 + wave) * 16;

    // Prefetch stage-1 region of packed W into cache while stage 0 runs
    // (gfx1250 global_prefetch_b8; 256 threads x 256B spread = 64KB).
    __builtin_prefetch(wpack + 32768 + ((size_t)threadIdx.x << 7), 0, 0);

    // --- preload 8 A fragments (16x32 f16 each; contiguous k order:
    // lane m=l&15, halves 0..15 -> K = kc+16h+e, matching packed B)
    const _Float16* arow = yp16 + ((size_t)b * T_ + t0 + lane16) * N_;
    v16h afrag[8];
#pragma unroll
    for (int kc = 0; kc < 8; ++kc) {
        const v8h lo = *reinterpret_cast<const v8h*>(arow + kc * 32 + 16 * h);
        const v8h hi = *reinterpret_cast<const v8h*>(arow + kc * 32 + 16 * h + 8);
#pragma unroll
        for (int e = 0; e < 8; ++e) {
            afrag[kc][e]     = lo[e];
            afrag[kc][8 + e] = hi[e];
        }
    }

    const uint4* wsrc = reinterpret_cast<const uint4*>(wpack);
    uint4*       wdst = reinterpret_cast<uint4*>(wlds);

    float rowacc[8] = {};
    for (int s = 0; s < 2; ++s) {
        __syncthreads();                   // protect LDS from previous stage
        // stage s: copy 64KB (tiles jt = s*8 .. s*8+7) into LDS, coalesced
        for (int i = threadIdx.x; i < 4096; i += 256)
            wdst[i] = wsrc[(size_t)s * 4096 + i];
        __syncthreads();

        for (int jh = 0; jh < 8; ++jh) {
            const int jt = s * 8 + jh;
            v8f acc = {};
#pragma unroll
            for (int kc = 0; kc < 8; ++kc) {
                const v16h bfrag = *reinterpret_cast<const v16h*>(
                    wlds + (((size_t)jh * 8 + kc) * 32 + lane) * 16);
                acc = __builtin_amdgcn_wmma_f32_16x16x32_f16(
                    false, afrag[kc], false, bfrag, (short)0, acc, false, false);
            }
            // acc VGPR r, lane l: Z[m = r + 8*(l>>4), j = jt*16 + (l&15)]
            const _Float16* ycol =
                yp16 + ((size_t)b * T_ + t0 + 8 * h) * N_ + jt * 16 + lane16;
#pragma unroll
            for (int r = 0; r < 8; ++r) {
                const float ypv = (float)ycol[(size_t)r * N_];
                rowacc[r] += (acc[r] + 1.0f) * ypv;  // +1 folds the Sum(y) term
            }
        }
    }

    // reduce across the 16 lanes holding the same row (masks < 16 keep halves)
#pragma unroll
    for (int r = 0; r < 8; ++r) {
        float v = rowacc[r];
        v += __shfl_xor(v, 1, 32);
        v += __shfl_xor(v, 2, 32);
        v += __shfl_xor(v, 4, 32);
        v += __shfl_xor(v, 8, 32);
        rowacc[r] = v;
    }
    if (lane16 == 0) {
#pragma unroll
        for (int r = 0; r < 8; ++r)
            opraw[(size_t)b * T_ + t0 + 8 * h + r] = rowacc[r] + BIAS_;
    }
}

// ---------------------------------------------------------------------------
// Kernel 4: sequential spike/reset scan, one lane per batch (16 chains).
// ---------------------------------------------------------------------------
__global__ void dbnn_scan(const float* __restrict__ opraw,
                          const float* __restrict__ tau_reset,
                          float* __restrict__ out) {
    const int b = threadIdx.x;
    if (b >= B_) return;
    const float d = __expf(-1.0f / tau_reset[0]);
    const float* raw = opraw + (size_t)b * T_;
    float*       o   = out   + (size_t)b * T_;

    float prev = raw[0];
    o[0] = prev;
    float r = 0.0f;
    for (int t = 1; t < T_; ++t) {
        const float rv = raw[t];
        r *= d;
        float v = rv + r;
        if (prev < THRES_ && v > THRES_) {
            const float k = ceilf((v - THRES_) / RESET_AMP_);
            r -= RESET_AMP_ * k;
            v = rv + r;
        }
        o[t] = v;
        prev = v;
    }
}

extern "C" void kernel_launch(void* const* d_in, const int* in_sizes, int n_in,
                              void* d_out, int out_size, void* d_ws, size_t ws_size,
                              hipStream_t stream) {
    const float* x         = (const float*)d_in[0];   // (B, N, T)
    const float* tau_reset = (const float*)d_in[1];   // (1,)
    const float* tau_rise  = (const float*)d_in[2];   // (N,)
    const float* tau_decay = (const float*)d_in[3];   // (N,)
    const float* omega     = (const float*)d_in[4];   // (N,)
    const float* W         = (const float*)d_in[5];   // (1, N, N)
    float* out = (float*)d_out;                       // (B, T)

    char* ws = (char*)d_ws;
    _Float16* yp16  = (_Float16*)ws;                                   // 16 MB
    _Float16* wpack = (_Float16*)(ws + (size_t)16 * 1024 * 1024);      // 128 KB
    float*    opraw = (float*)(ws + (size_t)16 * 1024 * 1024 + 131072);// 128 KB

    // 1) IIR depthwise conv, transposed f16 output
    dbnn_iir_conv<<<B_, N_, 0, stream>>>(x, tau_rise, tau_decay, omega, yp16);
    // 2) pack Wm into WMMA B-fragment layout (independent of kernel 1)
    dbnn_pack_w<<<128, 32, 0, stream>>>(W, wpack);
    // 3) bilinear + linear term via v_wmma_f32_16x16x32_f16, LDS-staged B
    dbnn_bilinear_wmma<<<dim3(B_, T_ / 128), 256, 0, stream>>>(yp16, wpack, opraw);
    // 4) sequential reset scan
    dbnn_scan<<<1, 32, 0, stream>>>(opraw, tau_reset, out);
}
